// SepConvLSTM_15101105013343
// MI455X (gfx1250) — compile-verified, matrix operand order checked by reading, not run on hardware
//
#include <hip/hip_runtime.h>

// SepConvLSTM fused step kernel for gfx1250 (wave32, WMMA).
// Pointwise 1x1 conv (the dominant GEMM, 256x128 x 128x(B*H*W)) runs on
// v_wmma_f32_16x16x32_f16. BN scales folded into weights at prep time.

typedef __attribute__((ext_vector_type(16))) _Float16 v16h;
typedef __attribute__((ext_vector_type(8)))  float    v8f;

#define B_   8
#define T_   20
#define CIN  64
#define HID  64
#define H_   64
#define W_   64
#define CC   128
#define C4   256

__device__ __forceinline__ float selu_f(float v) {
  const float lam = 1.0507009873554805f, alp = 1.6732632423543773f;
  return v > 0.f ? lam * v : lam * alp * (__expf(v) - 1.f);
}
__device__ __forceinline__ float sig_f(float v) { return 1.f / (1.f + __expf(-v)); }

// ---------------------------------------------------------------------------
// Prep: fold BN1 into depthwise weights/bias; fold BN2 scale into pointwise
// weights while packing them f16 in WMMA A-fragment order:
//   packed[((mt*4+kc)*32 + lane)*16 + j]  (mt: M-tile 0..15, kc: K-chunk 0..3)
//   M = mt*16 + (lane&15)
//   K = kc*32 + ((lane>>4)<<3) + ((j>>3)<<4) + (j&7)   [ISA 16-bit A 16x32 layout]
// ---------------------------------------------------------------------------
__global__ void sepconvlstm_prep(const float* __restrict__ dw_w,
                                 const float* __restrict__ g1, const float* __restrict__ be1,
                                 const float* __restrict__ m1, const float* __restrict__ v1,
                                 const float* __restrict__ pw_w,
                                 const float* __restrict__ g2, const float* __restrict__ be2,
                                 const float* __restrict__ m2, const float* __restrict__ v2,
                                 float* __restrict__ dww, float* __restrict__ b1,
                                 _Float16* __restrict__ pwp, float* __restrict__ bias2) {
  int tid = blockIdx.x * blockDim.x + threadIdx.x;
  if (tid < CC) {
    float s = g1[tid] * rsqrtf(v1[tid] + 1e-5f);
    b1[tid] = be1[tid] - m1[tid] * s;
#pragma unroll
    for (int k = 0; k < 9; ++k) dww[tid * 9 + k] = dw_w[tid * 9 + k] * s;
  }
  if (tid < C4) {
    float s = g2[tid] * rsqrtf(v2[tid] + 1e-5f);
    bias2[tid] = be2[tid] - m2[tid] * s;
  }
  if (tid < 16 * 4 * 32 * 16) {
    int j = tid & 15, l = (tid >> 4) & 31, kc = (tid >> 9) & 3, mt = tid >> 11;
    int m = mt * 16 + (l & 15);
    int k = kc * 32 + ((l >> 4) << 3) + ((j >> 3) << 4) + (j & 7);
    float s = g2[m] * rsqrtf(v2[m] + 1e-5f);
    pwp[tid] = (_Float16)(pw_w[m * CC + k] * s);
  }
}

// ---------------------------------------------------------------------------
// One ConvLSTM time step. Grid: (H=64, B=8); block: 256 threads = 8 waves.
// Each workgroup handles one image row (b, y): 64 pixels x all channels.
// ---------------------------------------------------------------------------
__global__ void __launch_bounds__(256)
sepconvlstm_step(const float* __restrict__ x_in,       // (B,T,CIN,H,W)
                 float* __restrict__ out,              // (B,T,HID,H,W)  (also h state)
                 float* __restrict__ cbuf,             // (B,HID,H,W)
                 const float* __restrict__ dww,        // folded depthwise weights (CC,9)
                 const float* __restrict__ b1,         // folded BN1 bias (CC)
                 const _Float16* __restrict__ pwp,     // packed pointwise f16 (16*4*32*16)
                 const float* __restrict__ bias2,      // folded BN2 bias (C4)
                 int t) {
  // LDS holds the B-matrix (K=128 x N=64) in WMMA B-fragment order:
  // ldsB[((nt*4+kc)*32 + lane)*16 + j], one contiguous 32B chunk per lane.
  __shared__ __align__(32) _Float16 ldsB[4 * 4 * 32 * 16];   // 16 KB

  const int tid = threadIdx.x;
  const int y   = blockIdx.x;
  const int b   = blockIdx.y;

  // ---------------- Stage 1: depthwise 3x3 + BN1 + SELU -> LDS (f16) -------
  {
    const int x  = tid & 63;     // pixel in row, coalesced across lanes
    const int cg = tid >> 6;     // channel group 0..3 (32 channels each)
#pragma unroll 4
    for (int ci = 0; ci < 32; ++ci) {
      const int c = cg * 32 + ci;
      const float* src = nullptr;
      if (c < CIN) {
        src = x_in + (((size_t)(b * T_ + t) * CIN + c) * H_) * W_;
      } else if (t > 0) {
        src = out + (((size_t)(b * T_ + (t - 1)) * HID + (c - CIN)) * H_) * W_;
      }
      float acc = b1[c];
      if (src) {
        const float* wp = dww + c * 9;
#pragma unroll
        for (int dy = -1; dy <= 1; ++dy) {
          const int yy = y + dy;
          if (yy < 0 || yy >= H_) continue;
          const float* row = src + yy * W_;
          const float v0 = (x > 0)      ? row[x - 1] : 0.f;
          const float v1 = row[x];
          const float v2 = (x < W_ - 1) ? row[x + 1] : 0.f;
          acc += wp[(dy + 1) * 3 + 0] * v0 + wp[(dy + 1) * 3 + 1] * v1 +
                 wp[(dy + 1) * 3 + 2] * v2;
        }
      }
      const float v = selu_f(acc);
      // scatter (c,x) -> B-fragment slot (bijection)
      const int kc = c >> 5, kk = c & 31;
      const int lane_s = (((kk >> 3) & 1) << 4) | (x & 15);
      const int jj     = (((kk >> 4) & 1) << 3) | (kk & 7);
      const int nt     = x >> 4;
      ldsB[((nt * 4 + kc) * 32 + lane_s) * 16 + jj] = (_Float16)v;
    }
  }
  __syncthreads();

  // ---------------- Stage 2: pointwise GEMM on WMMA ------------------------
  const int wv   = tid >> 5;
  const int lane = tid & 31;
  const int nt   = wv & 3;   // N-tile (16 pixels)
  const int mh   = wv >> 2;  // hid-channel half (0: ch 0..31, 1: ch 32..63)

  // acc[gate][s]: gate 0..3 = i,f,o,g ; s selects 16-channel sub-tile.
  // M-tile index = gate*4 + mh*2 + s  -> each lane ends up holding i,f,o,g
  // for the SAME 16 hid channels (in-register LSTM update, no cross-wave).
  v8f acc[4][2] = {};
#pragma unroll
  for (int kc = 0; kc < 4; ++kc) {
    const v16h bf = *(const v16h*)&ldsB[((nt * 4 + kc) * 32 + lane) * 16];
#pragma unroll
    for (int gate = 0; gate < 4; ++gate) {
#pragma unroll
      for (int s = 0; s < 2; ++s) {
        const int mt = gate * 4 + mh * 2 + s;
        const v16h af = *(const v16h*)&pwp[((mt * 4 + kc) * 32 + lane) * 16];
        acc[gate][s] = __builtin_amdgcn_wmma_f32_16x16x32_f16(
            false, af, false, bf, (short)0, acc[gate][s], false, false);
      }
    }
  }

  // ---------------- Stage 3: BN2 bias + SELU + sigmoid + LSTM update -------
  const int xp = nt * 16 + (lane & 15);
  const int lh = lane >> 4;   // C/D layout: VGPR r holds M = r + lh*8
#pragma unroll
  for (int s = 0; s < 2; ++s) {
#pragma unroll
    for (int r = 0; r < 8; ++r) {
      const int ch = (mh * 2 + s) * 16 + lh * 8 + r;   // hid channel 0..63
      const float iv = sig_f(selu_f(acc[0][s][r] + bias2[ch]));
      const float fv = sig_f(selu_f(acc[1][s][r] + bias2[64 + ch]));
      const float ov = sig_f(selu_f(acc[2][s][r] + bias2[128 + ch]));
      const float gv = sig_f(selu_f(acc[3][s][r] + bias2[192 + ch]));
      const size_t ci = (((size_t)b * HID + ch) * H_ + y) * W_ + xp;
      const float cp = cbuf[ci];
      const float cn = fv * cp + iv * gv;
      cbuf[ci] = cn;
      out[(((size_t)(b * T_ + t) * HID + ch) * H_ + y) * W_ + xp] = ov * tanhf(cn);
    }
  }
}

// ---------------------------------------------------------------------------
extern "C" void kernel_launch(void* const* d_in, const int* in_sizes, int n_in,
                              void* d_out, int out_size, void* d_ws, size_t ws_size,
                              hipStream_t stream) {
  const float* x_in  = (const float*)d_in[0];
  const float* dw_w  = (const float*)d_in[1];
  const float* g1    = (const float*)d_in[2];
  const float* be1   = (const float*)d_in[3];
  const float* m1    = (const float*)d_in[4];
  const float* v1    = (const float*)d_in[5];
  const float* pw_w  = (const float*)d_in[6];
  const float* g2    = (const float*)d_in[7];
  const float* be2   = (const float*)d_in[8];
  const float* m2    = (const float*)d_in[9];
  const float* v2    = (const float*)d_in[10];
  float* out = (float*)d_out;

  // workspace layout
  char* ws = (char*)d_ws;
  const size_t cbuf_bytes = (size_t)B_ * HID * H_ * W_ * sizeof(float);  // 8.39 MB
  float*    cbuf  = (float*)ws;                 ws += cbuf_bytes;
  _Float16* pwp   = (_Float16*)ws;              ws += (size_t)16 * 4 * 32 * 16 * sizeof(_Float16);
  float*    dww   = (float*)ws;                 ws += (size_t)CC * 9 * sizeof(float);
  float*    b1    = (float*)ws;                 ws += (size_t)CC * sizeof(float);
  float*    bias2 = (float*)ws;                 ws += (size_t)C4 * sizeof(float);
  (void)ws_size; (void)in_sizes; (void)n_in; (void)out_size;

  hipMemsetAsync(cbuf, 0, cbuf_bytes, stream);  // c0 = 0 (h0 handled via t==0 path)

  sepconvlstm_prep<<<(16 * 4 * 32 * 16 + 255) / 256, 256, 0, stream>>>(
      dw_w, g1, be1, m1, v1, pw_w, g2, be2, m2, v2, dww, b1, pwp, bias2);

  for (int t = 0; t < T_; ++t) {
    sepconvlstm_step<<<dim3(H_, B_), 256, 0, stream>>>(
        x_in, out, cbuf, dww, b1, pwp, bias2, t);
  }
}
